// GroupAttention_position_lepe_local_50714973831578
// MI455X (gfx1250) — matrix operand order repaired and checked by reading, not compiled, not run on hardware
//
#include <hip/hip_runtime.h>
#include <hip/hip_bf16.h>

typedef __attribute__((ext_vector_type(16))) _Float16 v16h;
typedef __attribute__((ext_vector_type(8)))  _Float16 v8h;
typedef __attribute__((ext_vector_type(8)))  float    v8f;

#define WMMA_F16(A, B, C) __builtin_amdgcn_wmma_f32_16x16x32_f16( \
    false, (A), false, (B), (short)0, (C), false, false)

// ---------------------------------------------------------------------------
// CDNA5 async global->LDS copy (16B per lane), tracked by ASYNCcnt.
// LDS dest passed as addrspace(3) pointer -> raw 32-bit LDS offset in a VGPR.
// ---------------------------------------------------------------------------
__device__ __forceinline__ void async_copy_b128(void* lds_dst,
                                                const void* gsrc) {
  auto ldsp = (__attribute__((address_space(3))) char*)lds_dst;
  asm volatile("global_load_async_to_lds_b128 %0, %1, off"
               :
               : "v"(ldsp), "v"(gsrc)
               : "memory");
}

__device__ __forceinline__ void wait_async0() {
  asm volatile("s_wait_asynccnt 0" ::: "memory");
}

// ---------------------------------------------------------------------------
// Generic f16 WMMA GEMM: C[M,N] = A[M,K] * B[K,N] (+bias)
// Block tile 128x128, 256 threads = 8 waves (2x4), wave tile 64x32 (4x2 accs).
// A f32 (convert while staging) or f16 (async-copied straight into LDS);
// B (weights) f32; out f16 or f32.
// M % 128 == 0, N % 128 == 0, K % 32 == 0 (true for all our shapes).
// ---------------------------------------------------------------------------
template <bool A_HALF, bool OUT_HALF, bool HAS_BIAS>
__global__ __launch_bounds__(256) void gemm_wmma_kernel(
    const void* __restrict__ Aptr, const float* __restrict__ Bptr,
    const float* __restrict__ bias, void* __restrict__ Cptr,
    int M, int N, int K) {
  __shared__ _Float16 As[128 * 40];   // [m][k], stride 40 halves (80 B)
  __shared__ _Float16 Bt[128 * 40];   // [n][k], stride 40 halves

  const int tid  = threadIdx.x;
  const int lane = tid & 31;
  const int wave = tid >> 5;       // 0..7
  const int wm   = wave >> 2;      // 0..1  (64 rows each)
  const int wn   = wave & 3;       // 0..3  (32 cols each)
  const int mBase = blockIdx.y * 128;
  const int nBase = blockIdx.x * 128;

  const float*    Af = (const float*)Aptr;
  const _Float16* Ah = (const _Float16*)Aptr;

  v8f acc[4][2] = {};

  const int kh  = lane >> 4;   // K-half select (0/1)
  const int l16 = lane & 15;

  for (int kc = 0; kc < K; kc += 32) {
    // ---- stage A tile 128x32 ----
    if (A_HALF) {
      // pure f16: async DMA straight to LDS, 16B chunks (4 per 32-half row)
      for (int idx = tid; idx < 512; idx += 256) {
        const int r  = idx >> 2;
        const int c8 = (idx & 3) * 8;
        async_copy_b128(&As[r * 40 + c8],
                        Ah + (size_t)(mBase + r) * K + kc + c8);
      }
    } else {
      for (int idx = tid; idx < 128 * 32; idx += 256) {
        const int r = idx >> 5, c = idx & 31;
        As[r * 40 + c] = (_Float16)Af[(size_t)(mBase + r) * K + kc + c];
      }
    }
    // ---- stage B tile 32x128 transposed (f32 -> f16) ----
    for (int idx = tid; idx < 32 * 128; idx += 256) {
      const int k = idx >> 7, n = idx & 127;
      Bt[n * 40 + k] = (_Float16)Bptr[(size_t)(kc + k) * N + nBase + n];
    }
    if (A_HALF) wait_async0();
    __syncthreads();

    // ---- fragments ----
    v16h afrag[4], bfrag[2];
#pragma unroll
    for (int mt = 0; mt < 4; ++mt) {
      const int m = wm * 64 + mt * 16 + l16;
      const v8h lo = *(const v8h*)&As[m * 40 + 8 * kh];        // K 8*kh..+7
      const v8h hi = *(const v8h*)&As[m * 40 + 16 + 8 * kh];   // K 16+8*kh..+7
      v16h f;
#pragma unroll
      for (int e = 0; e < 8; ++e) { f[e] = lo[e]; f[e + 8] = hi[e]; }
      afrag[mt] = f;
    }
#pragma unroll
    for (int nt = 0; nt < 2; ++nt) {
      const int n = wn * 32 + nt * 16 + l16;
      const v8h lo = *(const v8h*)&Bt[n * 40 + 8 * kh];        // K 8*kh..+7
      const v8h hi = *(const v8h*)&Bt[n * 40 + 16 + 8 * kh];   // K 16+8*kh..+7
      v16h f;
#pragma unroll
      for (int e = 0; e < 8; ++e) { f[e] = lo[e]; f[e + 8] = hi[e]; }
      bfrag[nt] = f;
    }
#pragma unroll
    for (int mt = 0; mt < 4; ++mt)
#pragma unroll
      for (int nt = 0; nt < 2; ++nt)
        acc[mt][nt] = WMMA_F16(afrag[mt], bfrag[nt], acc[mt][nt]);
    __syncthreads();
  }

  // ---- epilogue: C/D layout: VGPR r -> M=r (lanes 0-15) / r+8 (lanes 16-31)
  const int rowAdd = (lane & 16) ? 8 : 0;
#pragma unroll
  for (int mt = 0; mt < 4; ++mt)
#pragma unroll
    for (int nt = 0; nt < 2; ++nt)
#pragma unroll
      for (int r = 0; r < 8; ++r) {
        const int gm = mBase + wm * 64 + mt * 16 + r + rowAdd;
        const int gn = nBase + wn * 32 + nt * 16 + l16;
        float v = acc[mt][nt][r];
        if (HAS_BIAS) v += bias[gn];
        if (OUT_HALF)
          ((_Float16*)Cptr)[(size_t)gm * N + gn] = (_Float16)v;
        else
          ((float*)Cptr)[(size_t)gm * N + gn] = v;
      }
}

// ---------------------------------------------------------------------------
// Windowed attention + LePE. One wave per (batch, window, head).
// qkv: f16 [B*3136][1152] (q | k | v, 384 each). pre: f16 [B*3136][384].
// Q/K/V staged by async DMA; each wave waits only its own ASYNCcnt, so the
// kernel needs no barriers at all.
// ---------------------------------------------------------------------------
#define WPB 2  // waves per block

__global__ __launch_bounds__(32 * WPB) void attn_lepe_kernel(
    const _Float16* __restrict__ qkv, const float* __restrict__ lepe_w,
    const float* __restrict__ lepe_b, _Float16* __restrict__ pre) {
  __shared__ _Float16 QKVs[WPB][3][64 * 32];  // zero-padded 64 rows
  __shared__ float    Ssh[WPB][64 * 64];

  const int lane = threadIdx.x & 31;
  const int wv   = threadIdx.x >> 5;
  const long gw  = (long)blockIdx.x * WPB + wv;  // 0..24575
  const int h  = (int)(gw % 12);
  const int w  = (int)((gw / 12) % 64);
  const int b  = (int)(gw / (12 * 64));
  const int hy = w >> 3, wx = w & 7;

  _Float16* Qh = QKVs[wv][0];
  _Float16* Kh = QKVs[wv][1];
  _Float16* Vh = QKVs[wv][2];
  float*    S  = Ssh[wv];

  // zero S (padding rows/cols must be 0 for the padded WMMA tiles)
  for (int i = lane; i < 64 * 64; i += 32) S[i] = 0.0f;
  // zero Q/K/V padding rows 49..63
  for (int i = lane; i < 15 * 32; i += 32) {
    Qh[49 * 32 + i] = (_Float16)0.f;
    Kh[49 * 32 + i] = (_Float16)0.f;
    Vh[49 * 32 + i] = (_Float16)0.f;
  }

  // ---- async-stage Q/K/V rows 0..48: 49 tokens x 3 mats x 4 chunks(16B) ----
  const int cbase = h * 32;
  for (int cidx = lane; cidx < 49 * 12; cidx += 32) {
    const int s     = cidx / 12;
    const int rem   = cidx - s * 12;
    const int which = rem >> 2;        // 0=q, 1=k, 2=v
    const int c8    = (rem & 3) * 8;   // half offset within 32-ch row
    const int t     = (hy * 7 + s / 7) * 56 + wx * 7 + (s % 7);
    const _Float16* g =
        qkv + ((size_t)b * 3136 + t) * 1152 + which * 384 + cbase + c8;
    async_copy_b128(&QKVs[wv][which][s * 32 + c8], g);
  }
  wait_async0();  // this wave's DMA done; data visible to its own DS reads

  const int kh = lane >> 4, l16 = lane & 15;
  const int rowAdd = (lane & 16) ? 8 : 0;

  // ---- S = scale * Q @ K^T  (M=64pad, N=64pad, K=32: one WMMA per tile) ----
  v16h qf[4], kf[4];
#pragma unroll
  for (int mt = 0; mt < 4; ++mt) {
    const int m = mt * 16 + l16;
    const v8h lo = *(const v8h*)&Qh[m * 32 + 8 * kh];
    const v8h hi = *(const v8h*)&Qh[m * 32 + 16 + 8 * kh];
    v16h f;
#pragma unroll
    for (int e = 0; e < 8; ++e) { f[e] = lo[e]; f[e + 8] = hi[e]; }
    qf[mt] = f;
  }
#pragma unroll
  for (int nt = 0; nt < 4; ++nt) {  // B = K^T: row k=d, col n=token
    const int n = nt * 16 + l16;
    const v8h lo = *(const v8h*)&Kh[n * 32 + 8 * kh];
    const v8h hi = *(const v8h*)&Kh[n * 32 + 16 + 8 * kh];
    v16h f;
#pragma unroll
    for (int e = 0; e < 8; ++e) { f[e] = lo[e]; f[e + 8] = hi[e]; }
    kf[nt] = f;
  }

  const float scale = 0.17677669529663687f;  // 32^-0.5
#pragma unroll
  for (int mt = 0; mt < 4; ++mt)
#pragma unroll
    for (int nt = 0; nt < 4; ++nt) {
      v8f c = {};
      c = WMMA_F16(qf[mt], kf[nt], c);
#pragma unroll
      for (int r = 0; r < 8; ++r) {
        const int m = mt * 16 + r + rowAdd;
        const int n = nt * 16 + l16;
        if (m < 49 && n < 49) S[m * 64 + n] = c[r] * scale;
      }
    }

  // ---- softmax over valid 49 cols; padding stays zero ----
  for (int row = lane; row < 49; row += 32) {
    float* Sr = &S[row * 64];
    float mx = -1e30f;
    for (int n = 0; n < 49; ++n) mx = fmaxf(mx, Sr[n]);
    float sum = 0.f;
    for (int n = 0; n < 49; ++n) {
      const float e = __expf(Sr[n] - mx);
      Sr[n] = e;
      sum += e;
    }
    const float inv = 1.0f / sum;
    for (int n = 0; n < 49; ++n) Sr[n] *= inv;
  }

  // ---- O = P @ V  (M=64pad, N=32, K=64pad -> 2 chunks of 32) ----
  v8f oacc[4][2] = {};
#pragma unroll
  for (int kc = 0; kc < 2; ++kc) {
    v16h pf[4], vf[2];
#pragma unroll
    for (int mt = 0; mt < 4; ++mt) {
      const int m = mt * 16 + l16;
      v16h f;
#pragma unroll
      for (int e = 0; e < 16; ++e) {
        const int k = kc * 32 + (e < 8 ? 8 * kh + e : 16 + 8 * kh + (e - 8));
        f[e] = (_Float16)S[m * 64 + k];
      }
      pf[mt] = f;
    }
#pragma unroll
    for (int nt = 0; nt < 2; ++nt) {
      const int n = nt * 16 + l16;
      v16h f;
#pragma unroll
      for (int e = 0; e < 16; ++e) {
        const int k = kc * 32 + (e < 8 ? 8 * kh + e : 16 + 8 * kh + (e - 8));
        f[e] = Vh[k * 32 + n];
      }
      vf[nt] = f;
    }
#pragma unroll
    for (int mt = 0; mt < 4; ++mt)
#pragma unroll
      for (int nt = 0; nt < 2; ++nt)
        oacc[mt][nt] = WMMA_F16(pf[mt], vf[nt], oacc[mt][nt]);
  }

  // ---- O back to LDS (stride 32), P no longer needed ----
#pragma unroll
  for (int mt = 0; mt < 4; ++mt)
#pragma unroll
    for (int nt = 0; nt < 2; ++nt)
#pragma unroll
      for (int r = 0; r < 8; ++r) {
        const int m = mt * 16 + r + rowAdd;
        if (m < 49) S[m * 32 + nt * 16 + l16] = oacc[mt][nt][r];
      }

  // ---- LePE (depthwise 3x3, zero-pad per window) + fused output write ----
  const int c = cbase + lane;  // this lane's channel
  float w9[9];
#pragma unroll
  for (int i = 0; i < 9; ++i) w9[i] = lepe_w[c * 9 + i];
  const float bb = lepe_b[c];
  float vvv[49];
#pragma unroll
  for (int s = 0; s < 49; ++s) vvv[s] = (float)Vh[s * 32 + lane];

#pragma unroll
  for (int i = 0; i < 7; ++i)
#pragma unroll
    for (int j = 0; j < 7; ++j) {
      float a = bb;
#pragma unroll
      for (int di = -1; di <= 1; ++di)
#pragma unroll
        for (int dj = -1; dj <= 1; ++dj) {
          const int ii = i + di, jj = j + dj;
          if (ii >= 0 && ii < 7 && jj >= 0 && jj < 7)
            a += w9[(di + 1) * 3 + (dj + 1)] * vvv[ii * 7 + jj];
        }
      const int s = i * 7 + j;
      const float o = S[s * 32 + lane] + a;
      const int t = (hy * 7 + i) * 56 + wx * 7 + j;
      pre[((size_t)b * 3136 + t) * 384 + c] = (_Float16)o;
    }
}

// ---------------------------------------------------------------------------
extern "C" void kernel_launch(void* const* d_in, const int* in_sizes, int n_in,
                              void* d_out, int out_size, void* d_ws,
                              size_t ws_size, hipStream_t stream) {
  const float* x      = (const float*)d_in[0];  // [32,3136,384]
  const float* Wqkv   = (const float*)d_in[1];  // [384,1152]
  const float* Wproj  = (const float*)d_in[2];  // [384,384]
  const float* bproj  = (const float*)d_in[3];  // [384]
  const float* lepe_w = (const float*)d_in[4];  // [384,1,3,3]
  const float* lepe_b = (const float*)d_in[5];  // [384]
  float* out = (float*)d_out;

  const int Mtok = 32 * 3136;  // 100352

  // Workspace: qkv f16 [Mtok][1152] then pre f16 [Mtok][384]  (~308 MB)
  _Float16* qkv = (_Float16*)d_ws;
  _Float16* pre =
      (_Float16*)((char*)d_ws + (size_t)Mtok * 1152 * sizeof(_Float16));

  // 1) QKV projection: f32 in, f16 out
  gemm_wmma_kernel<false, true, false>
      <<<dim3(1152 / 128, Mtok / 128), dim3(256), 0, stream>>>(
          x, Wqkv, nullptr, qkv, Mtok, 1152, 384);

  // 2) Windowed attention + LePE (24576 window-heads, 1 wave each)
  attn_lepe_kernel<<<dim3(24576 / WPB), dim3(32 * WPB), 0, stream>>>(
      qkv, lepe_w, lepe_b, pre);

  // 3) Output projection: f16 in (async-staged), +bias, f32 out
  gemm_wmma_kernel<true, false, true>
      <<<dim3(384 / 128, Mtok / 128), dim3(256), 0, stream>>>(
          pre, Wproj, bproj, out, Mtok, 384, 384);
}
